// MultiHeadedAttention_68470368633226
// MI455X (gfx1250) — compile-verified, matrix operand order checked by reading, not run on hardware
//
#include <hip/hip_runtime.h>
#include <hip/hip_bf16.h>

// ---------------- fixed problem shape (from reference) ----------------
#define S_LEN   2048
#define B_SZ    2
#define D_MODEL 1024
#define NHEADS  16
#define DK      64
#define M_TOT   (B_SZ * S_LEN)   // 4096 rows for the projection GEMMs

typedef __attribute__((ext_vector_type(16))) __bf16 v16bf;
typedef __attribute__((ext_vector_type(8)))  float  v8f;

#define WMMA_BF16(a, b, c)                                                   \
    __builtin_amdgcn_wmma_f32_16x16x32_bf16(false, (a), false, (b),          \
                                            (short)0, (c), false, false)

static __device__ __forceinline__ unsigned short f2bf(float f) {
    unsigned u = __float_as_uint(f);
    u += 0x7FFFu + ((u >> 16) & 1u);          // round-to-nearest-even
    return (unsigned short)(u >> 16);
}

// Generic bf16 WMMA fragment loader for a K-contiguous row-major matrix.
// ISA 16-bit A/B layout (16x32): lane L holds line (line0 + L&15),
// K = 16*(e>>3) + 8*(L>>4) + (e&7)  -> two contiguous 16B chunks per lane.
static __device__ __forceinline__ v16bf load_frag(
        const unsigned short* __restrict__ base, int line0, int stride, int koff) {
    const int lane = threadIdx.x & 31;
    const unsigned short* p =
        base + (size_t)(line0 + (lane & 15)) * stride + koff + ((lane >> 4) << 3);
    union { uint4 u[2]; v16bf v; } t;
    t.u[0] = *(const uint4*)(p);        // K group 0: 8 bf16 (16B)
    t.u[1] = *(const uint4*)(p + 16);   // K group 1: 8 bf16 (16B)
    return t.v;
}

// ---------------- f32 -> bf16 conversion ----------------
__global__ void cvt_f32_bf16_kernel(const float* __restrict__ src,
                                    unsigned short* __restrict__ dst, int n) {
    int i = blockIdx.x * blockDim.x + threadIdx.x;
    if (i < n) dst[i] = f2bf(src[i]);
}

// ---------------- GEMM: C = A @ W^T + bias ----------------
// A: [M x K] bf16 row-major. W: [N x K] bf16 row-major (torch Linear weight).
// Wave tile 64x64 (16 WMMA per 32B-k-step) -> 32 flop/byte of fragment
// traffic, so the matrix pipe, not VMEM issue, is the limiter.
// mode 0: write bf16 heads layout  [B,H,S,DK]
// mode 1: write bf16 heads^T layout [B,H,DK,S]   (for V)
// mode 2: write f32 row-major [M,N]              (final projection -> d_out)
__global__ __launch_bounds__(128) void gemm_bf16_kernel(
    const unsigned short* __restrict__ A,
    const unsigned short* __restrict__ W,
    const float* __restrict__ bias,
    void* __restrict__ out,
    int M, int N, int K, int mode)
{
    const int lane = threadIdx.x & 31;
    const int wave = threadIdx.x >> 5;                 // 4 waves / block
    const int m0 = blockIdx.y * 128 + (wave >> 1) * 64;
    const int n0 = blockIdx.x * 128 + (wave & 1) * 64;

    v8f acc[4][4] = {};
    for (int k0 = 0; k0 < K; k0 += 32) {
        v16bf a[4], b[4];
#pragma unroll
        for (int i = 0; i < 4; ++i) a[i] = load_frag(A, m0 + 16 * i, K, k0);
#pragma unroll
        for (int j = 0; j < 4; ++j) b[j] = load_frag(W, n0 + 16 * j, K, k0);
#pragma unroll
        for (int i = 0; i < 4; ++i)
#pragma unroll
            for (int j = 0; j < 4; ++j)
                acc[i][j] = WMMA_BF16(a[i], b[j], acc[i][j]);
    }

    const int hf  = lane >> 4;     // which 16-lane half
    const int col = lane & 15;

    for (int i = 0; i < 4; ++i) {
        for (int j = 0; j < 4; ++j) {
            const int n  = n0 + 16 * j + col;
            const float bv = bias[n];
            for (int r = 0; r < 8; ++r) {
                const int m = m0 + 16 * i + r + 8 * hf;
                const float c = acc[i][j][r] + bv;
                if (mode == 2) {
                    ((float*)out)[(size_t)m * N + n] = c;
                } else {
                    const int b_ = m / S_LEN, s = m % S_LEN;
                    const int h  = n >> 6,    d = n & 63;
                    const size_t off = (mode == 0)
                        ? ((size_t)((b_ * NHEADS + h) * S_LEN + s) * DK + d)
                        : ((size_t)((b_ * NHEADS + h) * DK + d) * S_LEN + s);
                    ((unsigned short*)out)[off] = f2bf(c);
                }
            }
        }
    }
}

// ---------------- Flash attention ----------------
// One wave owns a 16-row q-tile of one (b,h). Streams keys 32 at a time:
//   S = Q Kc^T (2x 16x16 WMMA per 16 keys), online softmax, P via LDS
//   transpose into an A-fragment, O += P @ Vc (4 WMMA, dk split 4x16).
// Qh/Kh: [B,H,S,DK] bf16;  VhT: [B,H,DK,S] bf16;  Hout: [B,S,D] bf16.
__global__ __launch_bounds__(256) void flash_attn_kernel(
    const unsigned short* __restrict__ Qh,
    const unsigned short* __restrict__ Kh,
    const unsigned short* __restrict__ VhT,
    const int* __restrict__ mask,
    unsigned short* __restrict__ Hout)
{
    __shared__ float sP[8][16 * 32];     // per-wave P transpose buffer (16KB)

    const int lane = threadIdx.x & 31;
    const int wave = threadIdx.x >> 5;
    const int bh   = blockIdx.x;                 // 0..31
    const int b    = bh >> 4, h = bh & 15;
    const int q0   = blockIdx.y * 128 + wave * 16;
    const int hf   = lane >> 4;
    const int col  = lane & 15;

    const unsigned short* Qp = Qh  + (size_t)bh * S_LEN * DK;
    const unsigned short* Kp = Kh  + (size_t)bh * S_LEN * DK;
    const unsigned short* Vp = VhT + (size_t)bh * DK * S_LEN;
    const int*            Mp = mask + (size_t)b * S_LEN * S_LEN;

    const v16bf qa0 = load_frag(Qp, q0, DK, 0);
    const v16bf qa1 = load_frag(Qp, q0, DK, 32);

    v8f acc[4] = {};                     // O tile 16 x 64 (4 x 16-col groups)
    float mrun[8], lrun[8];
    for (int r = 0; r < 8; ++r) { mrun[r] = -1e30f; lrun[r] = 0.0f; }

    for (int k0 = 0; k0 < S_LEN; k0 += 32) {
        float sc[2][8];
        for (int t = 0; t < 2; ++t) {
            v16bf kb0 = load_frag(Kp, k0 + 16 * t, DK, 0);
            v16bf kb1 = load_frag(Kp, k0 + 16 * t, DK, 32);
            v8f s = {};
            s = WMMA_BF16(qa0, kb0, s);
            s = WMMA_BF16(qa1, kb1, s);
            const int kcol = k0 + 16 * t + col;
            for (int r = 0; r < 8; ++r) {
                const int qrow = q0 + r + 8 * hf;
                const int mv = Mp[(size_t)qrow * S_LEN + kcol];
                sc[t][r] = mv ? s[r] * 0.125f : -1e9f;   // * 1/sqrt(64), mask
            }
        }

        // Online softmax row statistics (rows live across 16-lane halves).
        float alpha[8];
        for (int r = 0; r < 8; ++r) {
            float rm = fmaxf(sc[0][r], sc[1][r]);
            for (int o = 1; o < 16; o <<= 1) rm = fmaxf(rm, __shfl_xor(rm, o, 32));
            const float mnew = fmaxf(mrun[r], rm);
            alpha[r] = __expf(mrun[r] - mnew);
            float rs = 0.0f;
            for (int t = 0; t < 2; ++t) {
                const float p = __expf(sc[t][r] - mnew);
                sc[t][r] = p;
                rs += p;
            }
            for (int o = 1; o < 16; o <<= 1) rs += __shfl_xor(rs, o, 32);
            lrun[r] = lrun[r] * alpha[r] + rs;
            mrun[r] = mnew;
        }

        // C-layout (lane=column) -> A-layout (lane=row) through LDS.
        for (int t = 0; t < 2; ++t)
            for (int r = 0; r < 8; ++r)
                sP[wave][(r + 8 * hf) * 32 + 16 * t + col] = sc[t][r];
        __syncthreads();

        union { unsigned short s16[16]; v16bf v; } pu;
        for (int e = 0; e < 16; ++e) {
            const int Kx = 16 * (e >> 3) + 8 * hf + (e & 7);
            pu.s16[e] = f2bf(sP[wave][col * 32 + Kx]);   // row = lane&15
        }
        __syncthreads();

        for (int g = 0; g < 4; ++g) {
            for (int r = 0; r < 8; ++r) acc[g][r] *= alpha[r];
            v16bf vb = load_frag(Vp, 16 * g, S_LEN, k0);  // B: [32 keys x 16 dk]
            acc[g] = WMMA_BF16(pu.v, vb, acc[g]);
        }
    }

    // Normalize and scatter back to [B,S,D] bf16 (head-major columns).
    for (int g = 0; g < 4; ++g) {
        const int d = h * DK + 16 * g + col;
        for (int r = 0; r < 8; ++r) {
            const int s = q0 + r + 8 * hf;
            const float o = acc[g][r] / lrun[r];
            Hout[(size_t)(b * S_LEN + s) * D_MODEL + d] = f2bf(o);
        }
    }
}

// ---------------- host side ----------------
extern "C" void kernel_launch(void* const* d_in, const int* in_sizes, int n_in,
                              void* d_out, int out_size, void* d_ws, size_t ws_size,
                              hipStream_t stream) {
    (void)in_sizes; (void)n_in; (void)out_size; (void)ws_size;

    const float* q_f  = (const float*)d_in[0];
    const float* k_f  = (const float*)d_in[1];
    const float* v_f  = (const float*)d_in[2];
    const int*   mask = (const int*)  d_in[3];
    const float* Wq = (const float*)d_in[4];  const float* bq = (const float*)d_in[5];
    const float* Wk = (const float*)d_in[6];  const float* bk = (const float*)d_in[7];
    const float* Wv = (const float*)d_in[8];  const float* bv = (const float*)d_in[9];
    const float* Wo = (const float*)d_in[10]; const float* bo = (const float*)d_in[11];

    const size_t NX = (size_t)M_TOT * D_MODEL;        // 4,194,304
    const size_t NW = (size_t)D_MODEL * D_MODEL;      // 1,048,576

    char* ws = (char*)d_ws;
    unsigned short* Xq  = (unsigned short*)(ws + 0);
    unsigned short* Xk  = (unsigned short*)(ws +  8u * 1024 * 1024);
    unsigned short* Xv  = (unsigned short*)(ws + 16u * 1024 * 1024);
    unsigned short* Wqb = (unsigned short*)(ws + 24u * 1024 * 1024);
    unsigned short* Wkb = (unsigned short*)(ws + 26u * 1024 * 1024);
    unsigned short* Wvb = (unsigned short*)(ws + 28u * 1024 * 1024);
    unsigned short* Wob = (unsigned short*)(ws + 30u * 1024 * 1024);
    unsigned short* Qh  = (unsigned short*)(ws + 32u * 1024 * 1024);
    unsigned short* Kh  = (unsigned short*)(ws + 40u * 1024 * 1024);
    unsigned short* VhT = (unsigned short*)(ws + 48u * 1024 * 1024);
    unsigned short* Hb  = (unsigned short*)(ws + 56u * 1024 * 1024);

    const int CB = 256;
    cvt_f32_bf16_kernel<<<(NX + CB - 1) / CB, CB, 0, stream>>>(q_f, Xq, (int)NX);
    cvt_f32_bf16_kernel<<<(NX + CB - 1) / CB, CB, 0, stream>>>(k_f, Xk, (int)NX);
    cvt_f32_bf16_kernel<<<(NX + CB - 1) / CB, CB, 0, stream>>>(v_f, Xv, (int)NX);
    cvt_f32_bf16_kernel<<<(NW + CB - 1) / CB, CB, 0, stream>>>(Wq, Wqb, (int)NW);
    cvt_f32_bf16_kernel<<<(NW + CB - 1) / CB, CB, 0, stream>>>(Wk, Wkb, (int)NW);
    cvt_f32_bf16_kernel<<<(NW + CB - 1) / CB, CB, 0, stream>>>(Wv, Wvb, (int)NW);
    cvt_f32_bf16_kernel<<<(NW + CB - 1) / CB, CB, 0, stream>>>(Wo, Wob, (int)NW);

    dim3 gblk(128);
    dim3 ggrid(D_MODEL / 128, M_TOT / 128);  // (8, 32)
    gemm_bf16_kernel<<<ggrid, gblk, 0, stream>>>(Xq, Wqb, bq, Qh,
                                                 M_TOT, D_MODEL, D_MODEL, 0);
    gemm_bf16_kernel<<<ggrid, gblk, 0, stream>>>(Xk, Wkb, bk, Kh,
                                                 M_TOT, D_MODEL, D_MODEL, 0);
    gemm_bf16_kernel<<<ggrid, gblk, 0, stream>>>(Xv, Wvb, bv, VhT,
                                                 M_TOT, D_MODEL, D_MODEL, 1);

    dim3 agrid(B_SZ * NHEADS, S_LEN / 128);  // (32, 16)
    flash_attn_kernel<<<agrid, dim3(256), 0, stream>>>(Qh, Kh, VhT, mask, Hb);

    gemm_bf16_kernel<<<ggrid, gblk, 0, stream>>>(Hb, Wob, bo, d_out,
                                                 M_TOT, D_MODEL, D_MODEL, 2);
}
